// MultiHeadAtten_84335977824793
// MI455X (gfx1250) — matrix operand order repaired and checked by reading, not compiled, not run on hardware
//
#include <hip/hip_runtime.h>
#include <hip/hip_bf16.h>
#include <stdint.h>

// ---------------------------------------------------------------------------
// MultiHeadAttention forward for MI455X (gfx1250, wave32, WMMA).
// B=4, L=2048, D=512, H=8, hd=64.
// All GEMMs use v_wmma_f32_16x16x32_f16 (f16 A/B, f32 accumulate).
// attn (536 MB) is written exactly once (fused scores+mask+softmax).
// ---------------------------------------------------------------------------

#define NB   4
#define NL   2048
#define ND   512
#define NH   8
#define HD   64

typedef __attribute__((ext_vector_type(16))) _Float16 v16h;
typedef __attribute__((ext_vector_type(8)))  float    v8f;

struct FragH {
  union { v16h h; uint32_t u[8]; };
};

__device__ __forceinline__ uint32_t pack2f16(float a, float b) {
  auto p = __builtin_amdgcn_cvt_pkrtz(a, b);       // v_cvt_pk_rtz_f16_f32
  union { decltype(p) h; uint32_t u; } cv; cv.h = p; return cv.u;
}

// A-fragment (16x32 f16, MxK) half-pair base K for VGPR v, lane l
__device__ __forceinline__ int a_k0(int v, int kg) {
  return ((v >> 2) << 4) + (kg << 3) + ((v & 3) << 1);
}
// B-fragment (32x16 f16, KxN) half-pair base K for VGPR v, lane l
__device__ __forceinline__ int b_k0(int v, int kg) {
  return ((v >> 2) << 4) + (kg << 3) + ((v & 3) << 1);
}

// ---------------------------------------------------------------------------
// Kernel 1: QKV projection.  out = x @ W^T + b, converted to f16.
// grid = (M/64, N/64, 3), block = 128 (4 waves); wave computes 16x64 tile.
// z==0 -> Q [bh][l][d], z==1 -> K [bh][l][d], z==2 -> V transposed [bh][d][l].
// ---------------------------------------------------------------------------
__global__ void __launch_bounds__(128)
qkv_proj_kernel(const float* __restrict__ xq, const float* __restrict__ xk,
                const float* __restrict__ xv,
                const float* __restrict__ Wq, const float* __restrict__ Wk,
                const float* __restrict__ Wv,
                const float* __restrict__ bq, const float* __restrict__ bk,
                const float* __restrict__ bv,
                _Float16* __restrict__ q16, _Float16* __restrict__ k16,
                _Float16* __restrict__ vt16)
{
  const int z    = blockIdx.z;
  const float* X = (z == 0) ? xq : (z == 1) ? xk : xv;
  const float* W = (z == 0) ? Wq : (z == 1) ? Wk : Wv;
  const float* bias = (z == 0) ? bq : (z == 1) ? bk : bv;

  const int lane = threadIdx.x & 31;
  const int wave = threadIdx.x >> 5;
  const int m    = lane & 15;
  const int kg   = lane >> 4;
  const int mBase = (blockIdx.x * 4 + wave) * 16;   // row in [0, B*L)
  const int nBase = blockIdx.y * 64;                // output channel base

  const float* xrow = X + (size_t)(mBase + m) * ND;

  v8f acc[4] = {};
  for (int kk = 0; kk < ND; kk += 32) {
    FragH a;
#pragma unroll
    for (int v = 0; v < 8; ++v) {
      float2 f = *(const float2*)(xrow + kk + a_k0(v, kg));
      a.u[v] = pack2f16(f.x, f.y);
    }
#pragma unroll
    for (int nt = 0; nt < 4; ++nt) {
      const int n = nBase + nt * 16 + m;            // lane's output column
      FragH bf;
#pragma unroll
      for (int v = 0; v < 8; ++v) {
        float2 f = *(const float2*)(W + (size_t)n * ND + kk + b_k0(v, kg));
        bf.u[v] = pack2f16(f.x, f.y);
      }
      acc[nt] = __builtin_amdgcn_wmma_f32_16x16x32_f16(
          false, a.h, false, bf.h, (short)0, acc[nt], false, false);
    }
  }

#pragma unroll
  for (int nt = 0; nt < 4; ++nt) {
    const int n  = nBase + nt * 16 + m;
    const float bb = bias[n];
    const int h = n >> 6;       // head
    const int d = n & 63;       // channel within head
#pragma unroll
    for (int r = 0; r < 8; ++r) {
      const int gm   = mBase + (kg << 3) + r;
      const int b    = gm >> 11;          // / NL
      const int lpos = gm & (NL - 1);
      const float val = acc[nt][r] + bb;
      const size_t bh = (size_t)b * NH + h;
      if (z == 0)      q16 [(bh * NL + lpos) * HD + d] = (_Float16)val;
      else if (z == 1) k16 [(bh * NL + lpos) * HD + d] = (_Float16)val;
      else             vt16[(bh * HD + d) * NL + lpos] = (_Float16)val; // V^T
    }
  }
}

// ---------------------------------------------------------------------------
// Kernel 2: fused scores + mask + softmax, writing attn (f32) exactly once.
// grid = (L/16, B*H), block = 512 (16 waves).
// Block owns 16 q-rows x all 2048 k; wave w owns k in [w*128, w*128+128).
// S = (Q K^T)/8 + mask*(-1e30); row softmax via shfl_xor + LDS reductions.
// ---------------------------------------------------------------------------
__global__ void __launch_bounds__(512)
attn_softmax_kernel(const _Float16* __restrict__ q16,
                    const _Float16* __restrict__ k16,
                    const int* __restrict__ mask,
                    float* __restrict__ attn)
{
  const int lane  = threadIdx.x & 31;
  const int wave  = threadIdx.x >> 5;     // 0..15
  const int bh    = blockIdx.y;
  const int b     = bh >> 3;
  const int qBase = blockIdx.x * 16;
  const int m     = lane & 15;
  const int kg    = lane >> 4;

  // Q fragments for the 16 q-rows (d = 0..63 -> two WMMA k-steps)
  const _Float16* qrow = q16 + ((size_t)bh * NL + qBase + m) * HD;
  FragH a0, a1;
#pragma unroll
  for (int v = 0; v < 8; ++v) {
    const int d = a_k0(v, kg);
    a0.u[v] = *(const uint32_t*)(qrow + d);
    a1.u[v] = *(const uint32_t*)(qrow + 32 + d);
  }

  v8f acc[8];
#pragma unroll
  for (int nt = 0; nt < 8; ++nt) {
    const int kcol = wave * 128 + nt * 16 + m;
    const _Float16* krow = k16 + ((size_t)bh * NL + kcol) * HD;
    FragH b0, b1;
#pragma unroll
    for (int v = 0; v < 8; ++v) {
      const int d = b_k0(v, kg);
      b0.u[v] = *(const uint32_t*)(krow + d);
      b1.u[v] = *(const uint32_t*)(krow + 32 + d);
    }
    v8f c = {};
    c = __builtin_amdgcn_wmma_f32_16x16x32_f16(false, a0.h, false, b0.h,
                                               (short)0, c, false, false);
    c = __builtin_amdgcn_wmma_f32_16x16x32_f16(false, a1.h, false, b1.h,
                                               (short)0, c, false, false);
    acc[nt] = c;
  }

  // scale + additive mask
#pragma unroll
  for (int nt = 0; nt < 8; ++nt) {
    const int kcol = wave * 128 + nt * 16 + m;
#pragma unroll
    for (int r = 0; r < 8; ++r) {
      const int q = qBase + (kg << 3) + r;
      const int mk = mask[((size_t)b * NL + q) * NL + kcol];
      acc[nt][r] = acc[nt][r] * 0.125f + (float)mk * (-1e30f);
    }
  }

  __shared__ float part[16][16];   // [wave][row]
  __shared__ float rowred[16];

  // ---- row max ----
  float mx[8];
#pragma unroll
  for (int r = 0; r < 8; ++r) {
    float v = acc[0][r];
#pragma unroll
    for (int nt = 1; nt < 8; ++nt) v = fmaxf(v, acc[nt][r]);
#pragma unroll
    for (int off = 1; off < 16; off <<= 1)      // reduce 16 lanes, same rows
      v = fmaxf(v, __shfl_xor(v, off, 32));
    mx[r] = v;
  }
  if ((lane & 15) == 0) {
#pragma unroll
    for (int r = 0; r < 8; ++r) part[wave][(kg << 3) + r] = mx[r];
  }
  __syncthreads();
  if (threadIdx.x < 16) {
    float v = part[0][threadIdx.x];
#pragma unroll
    for (int w = 1; w < 16; ++w) v = fmaxf(v, part[w][threadIdx.x]);
    rowred[threadIdx.x] = v;
  }
  __syncthreads();
  float rmax[8];
#pragma unroll
  for (int r = 0; r < 8; ++r) rmax[r] = rowred[(kg << 3) + r];
  __syncthreads();

  // ---- exp + row sum ----
  float ps[8];
#pragma unroll
  for (int r = 0; r < 8; ++r) ps[r] = 0.0f;
#pragma unroll
  for (int nt = 0; nt < 8; ++nt) {
#pragma unroll
    for (int r = 0; r < 8; ++r) {
      const float e = __expf(acc[nt][r] - rmax[r]);
      acc[nt][r] = e;
      ps[r] += e;
    }
  }
#pragma unroll
  for (int r = 0; r < 8; ++r) {
#pragma unroll
    for (int off = 1; off < 16; off <<= 1) ps[r] += __shfl_xor(ps[r], off, 32);
  }
  if ((lane & 15) == 0) {
#pragma unroll
    for (int r = 0; r < 8; ++r) part[wave][(kg << 3) + r] = ps[r];
  }
  __syncthreads();
  if (threadIdx.x < 16) {
    float v = part[0][threadIdx.x];
#pragma unroll
    for (int w = 1; w < 16; ++w) v += part[w][threadIdx.x];
    rowred[threadIdx.x] = v;
  }
  __syncthreads();
  float rinv[8];
#pragma unroll
  for (int r = 0; r < 8; ++r) rinv[r] = 1.0f / rowred[(kg << 3) + r];

  // ---- write normalized attn (single pass over the 536 MB output) ----
#pragma unroll
  for (int nt = 0; nt < 8; ++nt) {
    const int kcol = wave * 128 + nt * 16 + m;
#pragma unroll
    for (int r = 0; r < 8; ++r) {
      const int q = qBase + (kg << 3) + r;
      attn[((size_t)bh * NL + q) * NL + kcol] = acc[nt][r] * rinv[r];
    }
  }
}

// ---------------------------------------------------------------------------
// Kernel 3: out = attn @ V   ->  o16 [b][q][h*64+d]  (f16).
// grid = (L/64, B*H), block = 128 (4 waves); wave computes 16q x 64d,
// K-loop over 2048 attn columns (f32 -> f16 cvt_pk on the fly).
// ---------------------------------------------------------------------------
__global__ void __launch_bounds__(128)
attn_v_kernel(const float* __restrict__ attn,
              const _Float16* __restrict__ vt16,
              _Float16* __restrict__ o16)
{
  const int lane = threadIdx.x & 31;
  const int wave = threadIdx.x >> 5;
  const int bh   = blockIdx.y;
  const int b    = bh >> 3;
  const int h    = bh & 7;
  const int qBase = (blockIdx.x * 4 + wave) * 16;
  const int m    = lane & 15;
  const int kg   = lane >> 4;

  const float* arow = attn + ((size_t)bh * NL + qBase + m) * NL;

  v8f acc[4] = {};
  for (int kk = 0; kk < NL; kk += 32) {
    FragH a;
#pragma unroll
    for (int v = 0; v < 8; ++v) {
      float2 f = *(const float2*)(arow + kk + a_k0(v, kg));
      a.u[v] = pack2f16(f.x, f.y);
    }
#pragma unroll
    for (int nt = 0; nt < 4; ++nt) {
      const int d = nt * 16 + m;
      const _Float16* vrow = vt16 + ((size_t)bh * HD + d) * NL;  // V^T row
      FragH bf;
#pragma unroll
      for (int v = 0; v < 8; ++v)
        bf.u[v] = *(const uint32_t*)(vrow + kk + b_k0(v, kg));
      acc[nt] = __builtin_amdgcn_wmma_f32_16x16x32_f16(
          false, a.h, false, bf.h, (short)0, acc[nt], false, false);
    }
  }

#pragma unroll
  for (int nt = 0; nt < 4; ++nt) {
    const int d = nt * 16 + m;
#pragma unroll
    for (int r = 0; r < 8; ++r) {
      const int q = qBase + (kg << 3) + r;
      o16[((size_t)b * NL + q) * ND + h * HD + d] = (_Float16)acc[nt][r];
    }
  }
}

// ---------------------------------------------------------------------------
// Kernel 4: final = o @ Wo^T + bo  (f32 output).
// grid = (M/64, N/64), block = 128 (4 waves); wave computes 16x64 tile.
// ---------------------------------------------------------------------------
__global__ void __launch_bounds__(128)
out_proj_kernel(const _Float16* __restrict__ o16,
                const float* __restrict__ Wo, const float* __restrict__ bo,
                float* __restrict__ outF)
{
  const int lane = threadIdx.x & 31;
  const int wave = threadIdx.x >> 5;
  const int m    = lane & 15;
  const int kg   = lane >> 4;
  const int mBase = (blockIdx.x * 4 + wave) * 16;
  const int nBase = blockIdx.y * 64;

  const _Float16* arow = o16 + (size_t)(mBase + m) * ND;

  v8f acc[4] = {};
  for (int kk = 0; kk < ND; kk += 32) {
    FragH a;
#pragma unroll
    for (int v = 0; v < 8; ++v)
      a.u[v] = *(const uint32_t*)(arow + kk + a_k0(v, kg));
#pragma unroll
    for (int nt = 0; nt < 4; ++nt) {
      const int n = nBase + nt * 16 + m;
      FragH bf;
#pragma unroll
      for (int v = 0; v < 8; ++v) {
        float2 f = *(const float2*)(Wo + (size_t)n * ND + kk + b_k0(v, kg));
        bf.u[v] = pack2f16(f.x, f.y);
      }
      acc[nt] = __builtin_amdgcn_wmma_f32_16x16x32_f16(
          false, a.h, false, bf.h, (short)0, acc[nt], false, false);
    }
  }

#pragma unroll
  for (int nt = 0; nt < 4; ++nt) {
    const int n = nBase + nt * 16 + m;
    const float bb = bo[n];
#pragma unroll
    for (int r = 0; r < 8; ++r) {
      const int gm = mBase + (kg << 3) + r;
      outF[(size_t)gm * ND + n] = acc[nt][r] + bb;
    }
  }
}

// ---------------------------------------------------------------------------
extern "C" void kernel_launch(void* const* d_in, const int* in_sizes, int n_in,
                              void* d_out, int out_size, void* d_ws, size_t ws_size,
                              hipStream_t stream) {
  const float* xq = (const float*)d_in[0];
  const float* xk = (const float*)d_in[1];
  const float* xv = (const float*)d_in[2];
  const int*   mask = (const int*)d_in[3];
  const float* Wq = (const float*)d_in[4];
  const float* bq = (const float*)d_in[5];
  const float* Wk = (const float*)d_in[6];
  const float* bk = (const float*)d_in[7];
  const float* Wv = (const float*)d_in[8];
  const float* bv = (const float*)d_in[9];
  const float* Wo = (const float*)d_in[10];
  const float* bo = (const float*)d_in[11];

  float* outF = (float*)d_out;                                   // [B,L,512]
  float* attn = outF + (size_t)NB * NL * ND;                     // [B,H,L,L]

  const size_t qkvElems = (size_t)NB * NH * NL * HD;             // 4 Mi halfs
  _Float16* q16  = (_Float16*)d_ws;
  _Float16* k16  = q16 + qkvElems;
  _Float16* vt16 = k16 + qkvElems;
  _Float16* o16  = vt16 + qkvElems;                              // [B,L,512]

  // K1: QKV projection (z: 0=Q, 1=K, 2=V-transposed)
  {
    dim3 grid(NB * NL / 64, ND / 64, 3), block(128);
    qkv_proj_kernel<<<grid, block, 0, stream>>>(xq, xk, xv, Wq, Wk, Wv,
                                                bq, bk, bv, q16, k16, vt16);
  }
  // K2: fused scores + mask + softmax -> attn (written once)
  {
    dim3 grid(NL / 16, NB * NH), block(512);
    attn_softmax_kernel<<<grid, block, 0, stream>>>(q16, k16, mask, attn);
  }
  // K3: attn @ V -> o16
  {
    dim3 grid(NL / 64, NB * NH), block(128);
    attn_v_kernel<<<grid, block, 0, stream>>>(attn, vt16, o16);
  }
  // K4: output projection -> final
  {
    dim3 grid(NB * NL / 64, ND / 64), block(128);
    out_proj_kernel<<<grid, block, 0, stream>>>(o16, Wo, bo, outF);
  }
}
